// P2NM_90529320665576
// MI455X (gfx1250) — matrix operand order repaired and checked by reading, not compiled, not run on hardware
//
#include <hip/hip_runtime.h>

// ---------------------------------------------------------------------------
// Multi-scale patch attention (CDNA5 / gfx1250, wave32)
//   theta/phi/g convs -> bf16 unfold (d padded 784->832) -> fused flash
//   attention with v_wmma_f32_16x16x32_bf16 -> fold -> 1x1 conv + residual
// ---------------------------------------------------------------------------

#define BATCH 2
#define CIN   64
#define HH    128
#define WW    128
#define ICH   16
#define KSZ   7
#define TEMP  10.0f
#define DP    832          // padded patch dim: 784 -> 832 = 26*32
#define DC    26           // K-chunks of 32 along DP
#define NBLK  26           // 16-wide output column blocks per half (2*26*16=832)
#define QTOT  (HH*WW)      // 16384 queries
#define L2E   1.44269504088896f

typedef __attribute__((ext_vector_type(16))) __bf16 v16bf;
typedef __attribute__((ext_vector_type(8)))  float  v8f;

union V32B { uint4 u[2]; v16bf v; };

__device__ __forceinline__ unsigned short f2bf(float f) {
  union { float f; unsigned int u; } t; t.f = f;
  unsigned int r = t.u + 0x7fffu + ((t.u >> 16) & 1u);   // round-nearest-even
  return (unsigned short)(r >> 16);
}

// butterfly reductions within each 16-lane half (rows of the WMMA C layout)
__device__ __forceinline__ float red_max16(float v) {
  int i;
  i = __builtin_amdgcn_ds_swizzle(__float_as_int(v), 0x041f); v = fmaxf(v, __int_as_float(i));
  i = __builtin_amdgcn_ds_swizzle(__float_as_int(v), 0x081f); v = fmaxf(v, __int_as_float(i));
  i = __builtin_amdgcn_ds_swizzle(__float_as_int(v), 0x101f); v = fmaxf(v, __int_as_float(i));
  i = __builtin_amdgcn_ds_swizzle(__float_as_int(v), 0x201f); v = fmaxf(v, __int_as_float(i));
  return v;
}
__device__ __forceinline__ float red_sum16(float v) {
  int i;
  i = __builtin_amdgcn_ds_swizzle(__float_as_int(v), 0x041f); v += __int_as_float(i);
  i = __builtin_amdgcn_ds_swizzle(__float_as_int(v), 0x081f); v += __int_as_float(i);
  i = __builtin_amdgcn_ds_swizzle(__float_as_int(v), 0x101f); v += __int_as_float(i);
  i = __builtin_amdgcn_ds_swizzle(__float_as_int(v), 0x201f); v += __int_as_float(i);
  return v;
}

// ---------------------------------------------------------------------------
// conv: out[b][oc][y][x] = bias[oc] + sum_{c,dy,dx} x[b][c][y*s+dy][x*s+dx]*w
// w layout (ICH, CIN, s, s). s==1 handles theta; s in {1,2,4} for phi/g.
// ---------------------------------------------------------------------------
__global__ void conv_s_kernel(const float* __restrict__ x, const float* __restrict__ w,
                              const float* __restrict__ bias, float* __restrict__ out,
                              int s, int h, int total)
{
  int idx = blockIdx.x * blockDim.x + threadIdx.x;
  if (idx >= total) return;
  int xx = idx % h; int t = idx / h;
  int yy = t % h;   t /= h;
  int oc = t % ICH; int b = t / ICH;
  float acc = bias[oc];
  const float* xb = x + (size_t)b * CIN * HH * WW;
  const float* wr = w + (size_t)oc * CIN * s * s;
  for (int c = 0; c < CIN; ++c)
    for (int dy = 0; dy < s; ++dy)
      for (int dx = 0; dx < s; ++dx)
        acc += xb[((size_t)c * HH + (yy * s + dy)) * WW + (xx * s + dx)] *
               wr[(c * s + dy) * s + dx];
  out[idx] = acc;
}

// ---------------------------------------------------------------------------
// position-major unfold: dst[b][q][d] (bf16, d padded to DP), d=(c*7+dy)*7+dx,
// SAME pad 3, optional scale (TEMP folded into K here).
// ---------------------------------------------------------------------------
__global__ void unfold_pm_kernel(const float* __restrict__ src, unsigned short* __restrict__ dst,
                                 int h, float scale, int total)
{
  int idx = blockIdx.x * blockDim.x + threadIdx.x;
  if (idx >= total) return;
  int d = idx % DP; int t = idx / DP;
  int q = t % (h * h); int b = t / (h * h);
  float v = 0.f;
  if (d < KSZ * KSZ * ICH) {
    int c = d / 49, rem = d % 49, dy = rem / 7, dx = rem % 7;
    int qy = q / h, qx = q % h;
    int yy = qy + dy - 3, xx = qx + dx - 3;
    if (yy >= 0 && yy < h && xx >= 0 && xx < h)
      v = scale * src[((size_t)(b * ICH + c) * h + yy) * h + xx];
  }
  dst[idx] = f2bf(v);
}

// d-major unfold (for V/G): dst[b][d][key] so WMMA B-operand reads are contiguous
__global__ void unfold_dm_kernel(const float* __restrict__ src, unsigned short* __restrict__ dst,
                                 int h, int total)
{
  int idx = blockIdx.x * blockDim.x + threadIdx.x;
  if (idx >= total) return;
  int Lk = h * h;
  int key = idx % Lk; int t = idx / Lk;
  int d = t % DP; int b = t / DP;
  float v = 0.f;
  if (d < KSZ * KSZ * ICH) {
    int c = d / 49, rem = d % 49, dy = rem / 7, dx = rem % 7;
    int ky = key / h, kx = key % h;
    int yy = ky + dy - 3, xx = kx + dx - 3;
    if (yy >= 0 && yy < h && xx >= 0 && xx < h)
      v = src[((size_t)(b * ICH + c) * h + yy) * h + xx];
  }
  dst[idx] = f2bf(v);
}

// ---------------------------------------------------------------------------
// Fused flash attention. 2 waves/block, each wave = 16 queries, independent.
// grid = (QTOT/32, BATCH, 2 halves). Each half accumulates 26 of the 52
// output 16-column blocks (O acc = 26 * v8f = 208 VGPRs).
// ---------------------------------------------------------------------------
__global__ __launch_bounds__(64, 1)
void attn_kernel(const unsigned short* __restrict__ Qb,   // [b][q][DP] bf16
                 const unsigned short* __restrict__ Kb,   // [b][key][DP] bf16 (pre-scaled by TEMP)
                 const unsigned short* __restrict__ Gt,   // [b][DP][key] bf16
                 float* __restrict__ Yv,                   // [b][q][DP] f32
                 int Lk)
{
  extern __shared__ char smem[];
  const int lane = threadIdx.x & 31;
  const int wv   = threadIdx.x >> 5;
  const int b    = blockIdx.y;
  const int half = blockIdx.z;
  const int q0   = (blockIdx.x * 2 + wv) * 16;
  const int grp  = lane >> 4;        // 0: lanes 0-15, 1: lanes 16-31
  const int ln   = lane & 15;

  // per-wave LDS: Q tile in A-operand layout (26 KB) + P staging (1 KB)
  unsigned short* ldsQ = (unsigned short*)(smem + (size_t)wv * 27648);
  unsigned short* ldsP = ldsQ + DC * 512;

  // ---- stage Q into A-operand layout: lane L holds row M=L%16,
  //      elements 0..7 = d0+grp*8.., 8..15 = d0+16+grp*8..  (ISA 16-bit A table)
  {
    const unsigned short* qrow = Qb + ((size_t)(b * QTOT + q0 + ln)) * DP + grp * 8;
#pragma unroll
    for (int kc = 0; kc < DC; ++kc) {
      uint4 a0 = *(const uint4*)(qrow + kc * 32);
      uint4 a1 = *(const uint4*)(qrow + kc * 32 + 16);
      uint4* dq = (uint4*)(ldsQ + kc * 512 + lane * 16);
      dq[0] = a0; dq[1] = a1;
    }
  }
  __syncthreads();

  v8f o[NBLK];
#pragma unroll
  for (int i = 0; i < NBLK; ++i)
#pragma unroll
    for (int r = 0; r < 8; ++r) o[i][r] = 0.f;
  float m[8], l[8];
#pragma unroll
  for (int r = 0; r < 8; ++r) { m[r] = -3.0e38f; l[r] = 0.f; }

  const int    nkb   = Lk >> 5;
  const size_t kbase = (size_t)b * Lk;
  const int    dbase = half * (NBLK * 16) + ln;

  for (int j = 0; j < nkb; ++j) {
    const int k0 = j << 5;
    if (j + 1 < nkb)
      __builtin_prefetch(Kb + (kbase + k0 + 32 + lane) * DP, 0, 1);

    // ---- S = Q * K^T for 2 subtiles of 16 keys (scores already * TEMP)
    v8f u[2];
#pragma unroll
    for (int t = 0; t < 2; ++t) {
      v8f acc;
#pragma unroll
      for (int r = 0; r < 8; ++r) acc[r] = 0.f;
      const unsigned short* kp = Kb + (kbase + k0 + t * 16 + ln) * DP + grp * 8;
#pragma unroll
      for (int kc = 0; kc < DC; ++kc) {
        V32B a, bb;
        const uint4* qa = (const uint4*)(ldsQ + kc * 512 + lane * 16);
        a.u[0]  = qa[0];  a.u[1] = qa[1];
        bb.u[0] = *(const uint4*)(kp + kc * 32);
        bb.u[1] = *(const uint4*)(kp + kc * 32 + 16);
        acc = __builtin_amdgcn_wmma_f32_16x16x32_bf16(false, a.v, false, bb.v,
                                                      (short)0, acc, false, false);
      }
      u[t] = acc;
    }

    // ---- online softmax (rows of C layout align with O accumulator slots)
    float alpha[8], p0[8], p1[8];
#pragma unroll
    for (int r = 0; r < 8; ++r) {
      float tm = red_max16(fmaxf(u[0][r], u[1][r]));
      float mn = fmaxf(m[r], tm);
      alpha[r] = exp2f((m[r] - mn) * L2E);
      p0[r] = exp2f((u[0][r] - mn) * L2E);
      p1[r] = exp2f((u[1][r] - mn) * L2E);
      float sm = red_sum16(p0[r] + p1[r]);
      l[r] = l[r] * alpha[r] + sm;
      m[r] = mn;
    }
#pragma unroll
    for (int i = 0; i < NBLK; ++i)
#pragma unroll
      for (int r = 0; r < 8; ++r) o[i][r] *= alpha[r];

    // ---- stage P (16x32) through LDS into A-operand layout, bf16
#pragma unroll
    for (int t = 0; t < 2; ++t)
#pragma unroll
      for (int r = 0; r < 8; ++r) {
        int mrow = r + grp * 8;               // row in C layout
        int k    = t * 16 + ln;               // key column
        int dl   = mrow + (((k >> 3) & 1) << 4);   // dest lane in A layout
        int e    = (k & 7) + ((k >> 4) << 3);      // dest element in A layout
        ldsP[dl * 16 + e] = f2bf(t ? p1[r] : p0[r]);
      }
    asm volatile("s_wait_dscnt 0" ::: "memory");   // same-wave LDS ordering
    V32B pa;
    {
      const uint4* pp = (const uint4*)(ldsP + lane * 16);
      pa.u[0] = pp[0]; pa.u[1] = pp[1];
    }
    asm volatile("" ::: "memory");

    // ---- O += P * G  (G d-major -> contiguous B-operand reads)
#pragma unroll
    for (int blk = 0; blk < NBLK; ++blk) {
      const int d = dbase + blk * 16;
      const unsigned short* gp = Gt + ((size_t)(b * DP + d)) * Lk + k0 + grp * 8;
      V32B bb;
      bb.u[0] = *(const uint4*)(gp);
      bb.u[1] = *(const uint4*)(gp + 16);
      o[blk] = __builtin_amdgcn_wmma_f32_16x16x32_bf16(false, pa.v, false, bb.v,
                                                       (short)0, o[blk], false, false);
    }
  }

  // ---- epilogue: divide by softmax sum, write Yv (C layout scatter)
#pragma unroll
  for (int blk = 0; blk < NBLK; ++blk) {
    const int d = dbase + blk * 16;
#pragma unroll
    for (int r = 0; r < 8; ++r) {
      int q = q0 + r + grp * 8;
      Yv[((size_t)(b * QTOT + q)) * DP + d] = o[blk][r] / l[r];
    }
  }
}

// ---------------------------------------------------------------------------
// fold (gather form, no atomics) + mask division.
// out[y,x] = sum_{dy,dx valid} yv[q=(y+3-dy, x+3-dx)][(c*7+dy)*7+dx] / count
// ---------------------------------------------------------------------------
__global__ void fold_kernel(const float* __restrict__ yv, float* __restrict__ foldbuf,
                            int scaleIdx, int total)
{
  int idx = blockIdx.x * blockDim.x + threadIdx.x;
  if (idx >= total) return;
  int xx = idx % WW; int t = idx / WW;
  int yy = t % HH;   t /= HH;
  int c = t % ICH;   int b = t / ICH;
  float acc = 0.f; int cnt = 0;
  for (int dy = 0; dy < KSZ; ++dy) {
    int qy = yy + 3 - dy; if (qy < 0 || qy >= HH) continue;
    for (int dx = 0; dx < KSZ; ++dx) {
      int qx = xx + 3 - dx; if (qx < 0 || qx >= WW) continue;
      int q = qy * WW + qx;
      acc += yv[((size_t)(b * QTOT + q)) * DP + (c * 7 + dy) * 7 + dx];
      cnt++;
    }
  }
  foldbuf[((size_t)(b * 48 + scaleIdx * ICH + c) * HH + yy) * WW + xx] = acc / (float)cnt;
}

// final 1x1 conv over concat(48ch) + residual
__global__ void final_kernel(const float* __restrict__ x, const float* __restrict__ w_out,
                             const float* __restrict__ b_out, const float* __restrict__ foldbuf,
                             float* __restrict__ out, int total)
{
  int idx = blockIdx.x * blockDim.x + threadIdx.x;
  if (idx >= total) return;
  int p = idx % (HH * WW); int t = idx / (HH * WW);
  int co = t % CIN; int b = t / CIN;
  float acc = b_out[co];
  const float* fb = foldbuf + (size_t)b * 48 * HH * WW;
#pragma unroll 8
  for (int j = 0; j < 48; ++j)
    acc += w_out[co * 48 + j] * fb[(size_t)j * HH * WW + p];
  out[idx] = acc + x[idx];
}

// ---------------------------------------------------------------------------
extern "C" void kernel_launch(void* const* d_in, const int* in_sizes, int n_in,
                              void* d_out, int out_size, void* d_ws, size_t ws_size,
                              hipStream_t stream)
{
  (void)in_sizes; (void)n_in; (void)out_size; (void)ws_size;
  // setup_inputs() dict order:
  const float* x       = (const float*)d_in[0];
  const float* w_theta = (const float*)d_in[1];
  const float* b_theta = (const float*)d_in[2];
  const float* w_phi[3] = { (const float*)d_in[3], (const float*)d_in[7],  (const float*)d_in[11] };
  const float* b_phi[3] = { (const float*)d_in[4], (const float*)d_in[8],  (const float*)d_in[12] };
  const float* w_g[3]   = { (const float*)d_in[5], (const float*)d_in[9],  (const float*)d_in[13] };
  const float* b_g[3]   = { (const float*)d_in[6], (const float*)d_in[10], (const float*)d_in[14] };
  const float* w_out = (const float*)d_in[15];
  const float* b_out = (const float*)d_in[16];
  float* out = (float*)d_out;

  char* ws = (char*)d_ws;
  size_t off = 0;
  auto carve = [&](size_t bytes) { char* p = ws + off; off += (bytes + 255) & ~(size_t)255; return p; };
  float*          theta   = (float*)         carve((size_t)BATCH * ICH * HH * WW * 4);
  float*          phi     = (float*)         carve((size_t)BATCH * ICH * HH * WW * 4);
  float*          gbuf    = (float*)         carve((size_t)BATCH * ICH * HH * WW * 4);
  unsigned short* Qb      = (unsigned short*)carve((size_t)BATCH * QTOT * DP * 2);
  unsigned short* Kb      = (unsigned short*)carve((size_t)BATCH * QTOT * DP * 2);
  unsigned short* Gt      = (unsigned short*)carve((size_t)BATCH * QTOT * DP * 2);
  float*          Yv      = (float*)         carve((size_t)BATCH * QTOT * DP * 4);
  float*          foldbuf = (float*)         carve((size_t)BATCH * 48 * HH * WW * 4);

  const int TB = 256;
  {
    int total = BATCH * ICH * HH * WW;
    conv_s_kernel<<<(total + TB - 1) / TB, TB, 0, stream>>>(x, w_theta, b_theta, theta, 1, HH, total);
  }
  {
    int total = BATCH * QTOT * DP;
    unfold_pm_kernel<<<(total + TB - 1) / TB, TB, 0, stream>>>(theta, Qb, HH, 1.0f, total);
  }
  for (int i = 0; i < 3; ++i) {
    int s = 1 << i;
    int hs = HH / s;
    int Lk = hs * hs;
    int totc = BATCH * ICH * hs * hs;
    conv_s_kernel<<<(totc + TB - 1) / TB, TB, 0, stream>>>(x, w_phi[i], b_phi[i], phi, s, hs, totc);
    conv_s_kernel<<<(totc + TB - 1) / TB, TB, 0, stream>>>(x, w_g[i], b_g[i], gbuf, s, hs, totc);
    int totu = BATCH * Lk * DP;
    unfold_pm_kernel<<<(totu + TB - 1) / TB, TB, 0, stream>>>(phi, Kb, hs, TEMP, totu);
    unfold_dm_kernel<<<(totu + TB - 1) / TB, TB, 0, stream>>>(gbuf, Gt, hs, totu);
    dim3 g(QTOT / 32, BATCH, 2);
    attn_kernel<<<g, 64, 2 * 27648, stream>>>(Qb, Kb, Gt, Yv, Lk);
    int totf = BATCH * ICH * HH * WW;
    fold_kernel<<<(totf + TB - 1) / TB, TB, 0, stream>>>(Yv, foldbuf, i, totf);
  }
  {
    int total = BATCH * CIN * HH * WW;
    final_kernel<<<(total + TB - 1) / TB, TB, 0, stream>>>(x, w_out, b_out, foldbuf, out, total);
  }
}
